// KStore_77300821393784
// MI455X (gfx1250) — matrix operand order repaired and checked by reading, not compiled, not run on hardware
//
#include <hip/hip_runtime.h>
#include <hip/hip_bf16.h>
#include <math.h>

// ---------------------------------------------------------------------------
// Cosine-similarity retrieval (query 1024x1024, keys 262144x1024, top-32,
// gather values 262144x1024) tuned for MI455X (gfx1250, wave32, WMMA).
//
// Pipeline:
//  1) norm_rows_f16: L2-normalize rows, emit f16 (queries -> 2 MiB, keys ->
//     512 MiB scratch).  Done once so the compute-bound GEMM reads pure f16.
//  2) sim_chunk_topk: per workgroup, 32 queries x 1024 keys similarity tile.
//     Each wave computes a 32x32 macro-tile as 2x2 v_wmma_f32_16x16x32_f16,
//     so every A (LDS) and B (L2) fragment feeds TWO WMMAs -> operand traffic
//     per FLOP halved vs. a 1x1 mapping (4 WMMAs per 4 ds_b128 + 4 glb_b128).
//     Sim chunk staged in LDS (192 KB dynamic), then per-row top-32 via wave
//     argmax rounds -> candidate (score,index) pairs in scratch.
//  3) merge_rescore_gather: bitonic-sort 8192 candidates/row, rescore the
//     top-64 in exact fp32 cosine, re-sort, emit top-32 scores and gather
//     the 32 value rows.
// ---------------------------------------------------------------------------

typedef __attribute__((ext_vector_type(16))) _Float16 v16h;
typedef __attribute__((ext_vector_type(4)))  _Float16 v4h;
typedef __attribute__((ext_vector_type(8)))  float    v8f;

#define B_DIM 1024
#define D_DIM 1024
#define C_DIM 262144
#define V_DIM 1024
#define K_SEL 32
#define TILE_M 32                      // queries per workgroup
#define CHUNK 1024                     // keys per workgroup
#define NCHUNKS (C_DIM / CHUNK)        // 256
#define TILE_N 256                     // 8 waves x 32 cols per inner tile
#define INNER_TILES (CHUNK / TILE_N)   // 4
#define CAND_PER_ROW (NCHUNKS * K_SEL) // 8192
#define NEG_INF (-__builtin_inff())

// ---------------- Kernel 1: row-wise L2 normalize fp32 -> f16 --------------
__global__ __launch_bounds__(256) void norm_rows_f16(
    const float* __restrict__ in, _Float16* __restrict__ out)
{
  __shared__ float red[8];
  const int row = blockIdx.x;
  const int tid = threadIdx.x;
  const float4* src = (const float4*)(in + (size_t)row * D_DIM);

  float ss = 0.f;
  for (int i = tid; i < D_DIM / 4; i += 256) {
    float4 v = src[i];
    ss += v.x * v.x + v.y * v.y + v.z * v.z + v.w * v.w;
  }
#pragma unroll
  for (int off = 16; off; off >>= 1) ss += __shfl_xor(ss, off, 32);
  if ((tid & 31) == 0) red[tid >> 5] = ss;
  __syncthreads();
  if (tid == 0) {
    float t = 0.f;
#pragma unroll
    for (int w = 0; w < 8; ++w) t += red[w];
    red[0] = t;
  }
  __syncthreads();
  const float inv = 1.0f / fmaxf(sqrtf(red[0]), 1e-12f);

  _Float16* dst = out + (size_t)row * D_DIM;
  for (int i = tid; i < D_DIM / 4; i += 256) {
    float4 v = src[i];
    v4h h;
    h[0] = (_Float16)(v.x * inv);
    h[1] = (_Float16)(v.y * inv);
    h[2] = (_Float16)(v.z * inv);
    h[3] = (_Float16)(v.w * inv);
    *(v4h*)(dst + i * 4) = h;
  }
}

// ------- Kernel 2: WMMA sim tile (32x1024, 2x2 blocked) + chunk top-32 -----
__global__ __launch_bounds__(256) void sim_chunk_topk(
    const _Float16* __restrict__ q16,
    const _Float16* __restrict__ k16,
    float* __restrict__ cand_score,
    int*   __restrict__ cand_idx)
{
  extern __shared__ char smem[];
  float*    sim    = (float*)smem;                                       // 32 x CHUNK f32 (128 KB)
  _Float16* q_tile = (_Float16*)(smem + TILE_M * CHUNK * sizeof(float)); // 32 x D f16 (64 KB)

  const int tid  = threadIdx.x;
  const int wave = tid >> 5;
  const int lane = tid & 31;
  const int half = lane >> 4;   // 0: lanes 0-15, 1: lanes 16-31
  const int sub  = lane & 15;
  const int mBase     = blockIdx.x * TILE_M;
  const int chunkBase = blockIdx.y * CHUNK;

  { // stage the 32-query f16 tile into LDS (shared by all 8 waves)
    const uint4* src = (const uint4*)(q16 + (size_t)mBase * D_DIM);
    uint4* dst = (uint4*)q_tile;
    for (int i = tid; i < TILE_M * D_DIM / 8; i += 256) dst[i] = src[i];
  }
  __syncthreads();

  // A fragments: lane holds row M = sub (sub-tile 0) and M = 16+sub (sub-tile 1)
  const _Float16* arow0 = q_tile + sub * D_DIM;
  const _Float16* arow1 = q_tile + (16 + sub) * D_DIM;

  for (int nt = 0; nt < INNER_TILES; ++nt) {
    const int nBase = chunkBase + nt * TILE_N + wave * 32;
    const _Float16* krow0 = k16 + (size_t)(nBase + sub) * D_DIM;
    const _Float16* krow1 = k16 + (size_t)(nBase + 16 + sub) * D_DIM;
    v8f c00 = {}, c01 = {}, c10 = {}, c11 = {};
    for (int kb = 0; kb < D_DIM; kb += 32) {
      // A 16x32 f16 frags: lanes<16 hold K={kb..+7, kb+16..+23},
      //                    lanes>=16 hold K={kb+8..+15, kb+24..+31}
      v16h a0, a1, b0, b1;
      *((uint4*)&a0)       = *(const uint4*)(arow0 + kb + half * 8);
      *(((uint4*)&a0) + 1) = *(const uint4*)(arow0 + kb + 16 + half * 8);
      *((uint4*)&a1)       = *(const uint4*)(arow1 + kb + half * 8);
      *(((uint4*)&a1) + 1) = *(const uint4*)(arow1 + kb + 16 + half * 8);
      // B 32x16 f16 frags: same K split as A (kb+8*half..+7, kb+16+8*half..+7)
      *((uint4*)&b0)       = *(const uint4*)(krow0 + kb + half * 8);
      *(((uint4*)&b0) + 1) = *(const uint4*)(krow0 + kb + 16 + half * 8);
      *((uint4*)&b1)       = *(const uint4*)(krow1 + kb + half * 8);
      *(((uint4*)&b1) + 1) = *(const uint4*)(krow1 + kb + 16 + half * 8);
      // 2x2 register blocking: each fragment feeds two WMMAs
      c00 = __builtin_amdgcn_wmma_f32_16x16x32_f16(false, a0, false, b0,
                                                   (short)0, c00, false, false);
      c01 = __builtin_amdgcn_wmma_f32_16x16x32_f16(false, a0, false, b1,
                                                   (short)0, c01, false, false);
      c10 = __builtin_amdgcn_wmma_f32_16x16x32_f16(false, a1, false, b0,
                                                   (short)0, c10, false, false);
      c11 = __builtin_amdgcn_wmma_f32_16x16x32_f16(false, a1, false, b1,
                                                   (short)0, c11, false, false);
    }
    // C/D layout: vgpr r -> row 8*half + r, col = sub (+16 for N sub-tile 1)
    const int col = nt * TILE_N + wave * 32 + sub;
#pragma unroll
    for (int r = 0; r < 8; ++r) {
      const int mrow = half * 8 + r;
      sim[mrow * CHUNK + col]             = c00[r];
      sim[mrow * CHUNK + col + 16]        = c01[r];
      sim[(16 + mrow) * CHUNK + col]      = c10[r];
      sim[(16 + mrow) * CHUNK + col + 16] = c11[r];
    }
  }
  __syncthreads();

  // per-row top-32 over the 1024-wide chunk: one wave per row, argmax rounds
  for (int rb = 0; rb < TILE_M; rb += 8) {
    const int row = rb + wave;
    float* srow = sim + row * CHUNK;
    for (int sel = 0; sel < K_SEL; ++sel) {
      float best = NEG_INF;
      int bi = 0;
      for (int c = lane; c < CHUNK; c += 32) {
        float v = srow[c];
        if (v > best || (v == best && c < bi)) { best = v; bi = c; }
      }
#pragma unroll
      for (int off = 16; off; off >>= 1) {
        float ob = __shfl_xor(best, off, 32);
        int   oi = __shfl_xor(bi, off, 32);
        if (ob > best || (ob == best && oi < bi)) { best = ob; bi = oi; }
      }
      if (lane == 0) {
        const size_t o = (((size_t)(mBase + row)) * NCHUNKS + blockIdx.y) * K_SEL + sel;
        cand_score[o] = best;
        cand_idx[o]   = chunkBase + bi;
        srow[bi] = NEG_INF;  // remove winner for the next round
      }
      __syncthreads();  // uniform across all 8 waves
    }
  }
}

// ------ Kernel 3: merge candidates, fp32 rescore top-64, gather values -----
__global__ __launch_bounds__(256) void merge_rescore_gather(
    const float* __restrict__ cand_score,
    const int*   __restrict__ cand_idx,
    const float* __restrict__ query,
    const float* __restrict__ keys,
    const float* __restrict__ values,
    float* __restrict__ out_ret,     // (B, 32, V)
    float* __restrict__ out_scores)  // (B, 32)
{
  extern __shared__ char smem[];
  float* s    = (float*)smem;                                  // 8192 f32
  int*   si   = (int*)(smem + CAND_PER_ROW * 4);               // 8192 i32
  float* qrow = (float*)(smem + CAND_PER_ROW * 8);             // 1024 f32
  float* red  = (float*)(smem + CAND_PER_ROW * 8 + D_DIM * 4); // 8
  float* rs   = red + 8;                                       // 64
  int*   ri   = (int*)(rs + 64);                               // 64

  const int b    = blockIdx.x;
  const int tid  = threadIdx.x;
  const int wave = tid >> 5;
  const int lane = tid & 31;

  for (int i = tid; i < CAND_PER_ROW; i += 256) {
    s[i]  = cand_score[(size_t)b * CAND_PER_ROW + i];
    si[i] = cand_idx  [(size_t)b * CAND_PER_ROW + i];
  }
  // stage fp32 query row + its sum of squares
  float ss = 0.f;
  for (int i = tid; i < D_DIM; i += 256) {
    float v = query[(size_t)b * D_DIM + i];
    qrow[i] = v;
    ss += v * v;
  }
#pragma unroll
  for (int off = 16; off; off >>= 1) ss += __shfl_xor(ss, off, 32);
  if (lane == 0) red[wave] = ss;
  __syncthreads();

  // bitonic sort 8192 candidates, descending score, index tiebreak
  for (int k = 2; k <= CAND_PER_ROW; k <<= 1) {
    for (int j = k >> 1; j > 0; j >>= 1) {
      for (int i = tid; i < CAND_PER_ROW; i += 256) {
        const int ixj = i ^ j;
        if (ixj > i) {
          float sa = s[i], sb = s[ixj];
          int   ia = si[i], ib = si[ixj];
          const bool aFirst  = (sa > sb) || (sa == sb && ia < ib);
          const bool descend = ((i & k) == 0);
          if (descend != aFirst) { s[i] = sb; s[ixj] = sa; si[i] = ib; si[ixj] = ia; }
        }
      }
      __syncthreads();
    }
  }

  if (tid == 0) {
    float t = 0.f;
#pragma unroll
    for (int w = 0; w < 8; ++w) t += red[w];
    red[0] = t;
  }
  __syncthreads();
  const float qinv = 1.0f / fmaxf(sqrtf(red[0]), 1e-12f);

  // exact fp32 cosine rescore of the top-64 candidates (one wave per cand)
  for (int j = wave; j < 64; j += 8) {
    const int kr_idx = si[j];
    const float* kr = keys + (size_t)kr_idx * D_DIM;
    float dot = 0.f, kss = 0.f;
    for (int i = lane; i < D_DIM; i += 32) {
      float kv = kr[i];
      dot += kv * qrow[i];
      kss += kv * kv;
    }
#pragma unroll
    for (int off = 16; off; off >>= 1) {
      dot += __shfl_xor(dot, off, 32);
      kss += __shfl_xor(kss, off, 32);
    }
    if (lane == 0) {
      const float kinv = 1.0f / fmaxf(sqrtf(kss), 1e-12f);
      rs[j] = dot * qinv * kinv;
      ri[j] = kr_idx;
    }
  }
  __syncthreads();

  // bitonic sort 64 rescored candidates (descending); first 32 = result
  for (int k = 2; k <= 64; k <<= 1) {
    for (int j = k >> 1; j > 0; j >>= 1) {
      if (tid < 64) {
        const int i = tid, ixj = i ^ j;
        if (ixj > i) {
          float sa = rs[i], sb = rs[ixj];
          int   ia = ri[i], ib = ri[ixj];
          const bool aFirst  = (sa > sb) || (sa == sb && ia < ib);
          const bool descend = ((i & k) == 0);
          if (descend != aFirst) { rs[i] = sb; rs[ixj] = sa; ri[i] = ib; ri[ixj] = ia; }
        }
      }
      __syncthreads();
    }
  }

  if (tid < K_SEL) out_scores[(size_t)b * K_SEL + tid] = rs[tid];

  // gather the 32 value rows (float4, fully coalesced)
  for (int t = tid; t < K_SEL * (V_DIM / 4); t += 256) {
    const int j = t / (V_DIM / 4);
    const int c = t % (V_DIM / 4);
    const int srcRow = ri[j];
    ((float4*)out_ret)[((size_t)b * K_SEL + j) * (V_DIM / 4) + c] =
        ((const float4*)values)[(size_t)srcRow * (V_DIM / 4) + c];
  }
}

// ---------------------------------------------------------------------------
extern "C" void kernel_launch(void* const* d_in, const int* in_sizes, int n_in,
                              void* d_out, int out_size, void* d_ws, size_t ws_size,
                              hipStream_t stream) {
  (void)in_sizes; (void)n_in; (void)out_size; (void)ws_size;
  const float* query  = (const float*)d_in[0];
  const float* keys   = (const float*)d_in[1];
  const float* values = (const float*)d_in[2];
  // d_in[3] holds k; fixed at 32 per the reference setup.

  char* ws = (char*)d_ws;
  _Float16* q16    = (_Float16*)ws;                            //   2 MiB
  _Float16* k16    = (_Float16*)(ws + ((size_t)2   << 20));    // 512 MiB
  float*    cand_s = (float*)   (ws + ((size_t)514 << 20));    //  32 MiB
  int*      cand_i = (int*)     (ws + ((size_t)546 << 20));    //  32 MiB

  float* out_ret = (float*)d_out;
  float* out_sc  = out_ret + (size_t)B_DIM * K_SEL * V_DIM;

  norm_rows_f16<<<dim3(B_DIM), dim3(256), 0, stream>>>(query, q16);
  norm_rows_f16<<<dim3(C_DIM), dim3(256), 0, stream>>>(keys, k16);

  // grid.x = M tiles (fast-varying) so consecutive blocks share the same
  // 2 MiB key chunk -> key reuse is captured by the 192 MB L2.
  dim3 g(B_DIM / TILE_M, NCHUNKS);
  const size_t lds_gemm = (size_t)TILE_M * CHUNK * sizeof(float) +
                          (size_t)TILE_M * D_DIM * sizeof(_Float16);  // 192 KB
  sim_chunk_topk<<<g, dim3(256), lds_gemm, stream>>>(q16, k16, cand_s, cand_i);

  const size_t lds_merge = (size_t)CAND_PER_ROW * 8 + (size_t)D_DIM * 4 +
                           8 * 4 + 64 * 4 + 64 * 4;  // ~68.5 KB
  merge_rescore_gather<<<dim3(B_DIM), dim3(256), lds_merge, stream>>>(
      cand_s, cand_i, query, keys, values, out_ret, out_sc);
}
